// MultiHeadLatentAttention_76802605187773
// MI455X (gfx1250) — matrix operand order repaired and checked by reading, not compiled, not run on hardware
//
#include <hip/hip_runtime.h>
#include <hip/hip_bf16.h>

// ---------------------------------------------------------------------------
// MLA block for MI455X (gfx1250): everything GEMM-shaped goes through
// v_wmma_f32_16x16x32_bf16 with f32 accumulation. Compute-bound problem
// (~150 GFLOP over ~60MB -> L2 resident), so we optimize for WMMA density and
// use the CDNA5 async global->LDS path (ASYNCcnt) to stage shared B tiles.
// ---------------------------------------------------------------------------

typedef __attribute__((ext_vector_type(16))) __bf16 v16bf;
typedef __attribute__((ext_vector_type(8)))  __bf16 v8bf;
typedef __attribute__((ext_vector_type(8)))  float  v8f;

#define NHEAD 16
#define HD    128
#define RDIM  64
#define HDIM  192   // HD + RDIM
#define TT    2048
#define DD    2048
#define QLAT  1536
#define KVLAT 512
#define NEGBIG (-1e30f)

// D = A(16x32) * B(32x16) + C, bf16 in / f32 out
__device__ __forceinline__ v8f wmma_bf16(v16bf a, v16bf b, v8f c) {
  return __builtin_amdgcn_wmma_f32_16x16x32_bf16(false, a, false, b, (short)0, c,
                                                 false, false);
}

// A fragment (16x32 bf16) from row-major [*, ld] matrix at (row, kc).
// Layout used here (contiguous K, same as B): lane m=L%16 g=L>>4; elem j -> K = kc+g*16+j
__device__ __forceinline__ v16bf load_a_frag(const __bf16* base, int ld, int row,
                                             int kc, int lane) {
  int m = lane & 15, g = lane >> 4;
  const __bf16* p = base + (size_t)(row + m) * (size_t)ld + kc;
  v8bf lo = *(const v8bf*)(p + g * 16);
  v8bf hi = *(const v8bf*)(p + g * 16 + 8);
  v16bf a;
#pragma unroll
  for (int i = 0; i < 8; i++) { a[i] = lo[i]; a[i + 8] = hi[i]; }
  return a;
}

// B fragment (32x16 bf16) taken from an N-major (transposed, row = column-of-B,
// contiguous K) matrix [*, ld] at (col, kc).
// Layout used here (contiguous K, same as A): lane n=L%16 g=L>>4; elem j -> K = kc+g*16+j
__device__ __forceinline__ v16bf load_b_frag(const __bf16* base, int ld, int col,
                                             int kc, int lane) {
  int n = lane & 15, g = lane >> 4;
  const __bf16* p = base + (size_t)(col + n) * (size_t)ld + kc + g * 16;
  v8bf lo = *(const v8bf*)(p);
  v8bf hi = *(const v8bf*)(p + 8);
  v16bf b;
#pragma unroll
  for (int i = 0; i < 8; i++) { b[i] = lo[i]; b[i + 8] = hi[i]; }
  return b;
}

// CDNA5 async global->LDS copy of 16 bytes per lane (ASYNCcnt-tracked).
// VDST VGPR carries the LDS byte address (low 32 bits of the generic pointer,
// per the flat-LDS aperture truncation rule).
__device__ __forceinline__ void async_copy_b128(unsigned lds_addr, const void* gsrc) {
  asm volatile("global_load_async_to_lds_b128 %0, %1, off"
               :: "v"(lds_addr), "v"(gsrc) : "memory");
}

// ---------------------------------------------------------------------------
// Generic GEMM: C[MxN] = A[MxK] * B[KxN], with B supplied transposed (Bt: NxK).
// Block = 256 threads = 8 waves; block tile 256(M) x 64(N); wave = 32x64 strip
// (2 A frags x 4 B frags -> 8 WMMAs per 32-K chunk). The shared 64x32 Bt tile
// is staged through LDS with double-buffered async copies.
// ---------------------------------------------------------------------------
__global__ __launch_bounds__(256) void gemm_bf16_nt(const __bf16* __restrict__ A,
                                                    const __bf16* __restrict__ Bt,
                                                    float* __restrict__ C,
                                                    int M, int N, int K) {
  __shared__ __bf16 Bs[2][64][32];
  int tid = threadIdx.x;
  int lane = tid & 31, wave = tid >> 5;
  int mBase = blockIdx.y * 256 + wave * 32;
  int nBase = blockIdx.x * 64;

  // per-thread async staging slot: thread copies 16B of Bt tile
  int scol = tid >> 2;          // 0..63 : column of B (row of Bt tile)
  int sch  = (tid & 3) * 8;     // 0,8,16,24 : K sub-chunk
  const __bf16* gsrc = Bt + (size_t)(nBase + scol) * (size_t)K + sch;
  unsigned lds0 = (unsigned)(size_t)(&Bs[0][scol][sch]);
  unsigned lds1 = (unsigned)(size_t)(&Bs[1][scol][sch]);

  // prologue: stage K-chunk 0 into buffer 0
  async_copy_b128(lds0, gsrc);

  v8f acc0[4] = {};
  v8f acc1[4] = {};
  int bufsel = 0;
  for (int kc = 0; kc < K; kc += 32) {
    if (kc + 32 < K) {
      // prefetch next chunk into the other buffer, then wait for current
      async_copy_b128(bufsel ? lds0 : lds1, gsrc + kc + 32);
      asm volatile("s_wait_asynccnt 0x1" ::: "memory");
    } else {
      asm volatile("s_wait_asynccnt 0x0" ::: "memory");
    }
    __syncthreads();  // all waves' copies for current buffer have landed

    v16bf a0 = load_a_frag(A, K, mBase, kc, lane);
    v16bf a1 = load_a_frag(A, K, mBase + 16, kc, lane);
    {
      int n16 = lane & 15, g = lane >> 4;
#pragma unroll
      for (int n = 0; n < 4; n++) {
        const __bf16* p = &Bs[bufsel][n * 16 + n16][g * 16];
        v8bf lo = *(const v8bf*)(p);
        v8bf hi = *(const v8bf*)(p + 8);
        v16bf b;
#pragma unroll
        for (int i = 0; i < 8; i++) { b[i] = lo[i]; b[i + 8] = hi[i]; }
        acc0[n] = wmma_bf16(a0, b, acc0[n]);
        acc1[n] = wmma_bf16(a1, b, acc1[n]);
      }
    }
    __syncthreads();  // everyone done reading before buffer is overwritten
    bufsel ^= 1;
  }

  int g = lane >> 4, cn = lane & 15;
#pragma unroll
  for (int n = 0; n < 4; n++) {
#pragma unroll
    for (int v = 0; v < 8; v++) {
      C[(size_t)(mBase + v + 8 * g) * (size_t)N + nBase + n * 16 + cn] = acc0[n][v];
      C[(size_t)(mBase + 16 + v + 8 * g) * (size_t)N + nBase + n * 16 + cn] = acc1[n][v];
    }
  }
}

// ---------------------------------------------------------------------------
// Elementwise fp32 -> bf16
// ---------------------------------------------------------------------------
__global__ void to_bf16_kernel(const float* __restrict__ in, __bf16* __restrict__ out,
                               size_t n) {
  size_t i = (size_t)blockIdx.x * blockDim.x + threadIdx.x;
  if (i < n) out[i] = (__bf16)in[i];
}

// fp32 KxN -> bf16 NxK (transpose during precision convert)
__global__ void transpose_to_bf16(const float* __restrict__ in, __bf16* __restrict__ out,
                                  int K, int N) {
  size_t idx = (size_t)blockIdx.x * blockDim.x + threadIdx.x;
  size_t total = (size_t)K * N;
  if (idx < total) {
    size_t k = idx / N, n = idx % N;
    out[n * (size_t)K + k] = (__bf16)in[idx];
  }
}

// ---------------------------------------------------------------------------
// RMSNorm over rows of width N, fp32 in, bf16 out (with weight)
// ---------------------------------------------------------------------------
__global__ __launch_bounds__(256) void rmsnorm_bf16(const float* __restrict__ in,
                                                    const float* __restrict__ w,
                                                    __bf16* __restrict__ out, int N) {
  __shared__ float red[256];
  int row = blockIdx.x;
  const float* r = in + (size_t)row * N;
  float ss = 0.f;
  for (int i = threadIdx.x; i < N; i += 256) { float v = r[i]; ss += v * v; }
  red[threadIdx.x] = ss;
  __syncthreads();
  for (int s = 128; s > 0; s >>= 1) {
    if (threadIdx.x < s) red[threadIdx.x] += red[threadIdx.x + s];
    __syncthreads();
  }
  float scale = rsqrtf(red[0] / (float)N + 1e-6f);
  for (int i = threadIdx.x; i < N; i += 256)
    out[(size_t)row * N + i] = (__bf16)(r[i] * scale * w[i]);
}

// ---------------------------------------------------------------------------
// alpha = sigmoid(x @ w_gate + b_gate), one block per token
// ---------------------------------------------------------------------------
__global__ __launch_bounds__(256) void gate_kernel(const float* __restrict__ x,
                                                   const float* __restrict__ wg,
                                                   const float* __restrict__ bg,
                                                   float* __restrict__ alpha,
                                                   float* __restrict__ outA, int D) {
  __shared__ float red[256];
  int t = blockIdx.x;
  const float* r = x + (size_t)t * D;
  float s = 0.f;
  for (int i = threadIdx.x; i < D; i += 256) s += r[i] * wg[i];
  red[threadIdx.x] = s;
  __syncthreads();
  for (int q = 128; q > 0; q >>= 1) {
    if (threadIdx.x < q) red[threadIdx.x] += red[threadIdx.x + q];
    __syncthreads();
  }
  if (threadIdx.x == 0) {
    float a = 1.f / (1.f + __expf(-(red[0] + bg[0])));
    alpha[t] = a;
    outA[t] = a;
  }
}

// ---------------------------------------------------------------------------
// RoPE (uncertainty-scaled) + assemble per-head tensors:
//   Q[h][t][0..191]  (row-major per head; A operand for scores)
//   K[h][t][0..191]  (row-major per head; doubles as Bt operand for Q.K^T)
//   Vt[h][d][t]      (d-major; Bt operand for P.V)
// grid (T, NH), block 192
// ---------------------------------------------------------------------------
__global__ __launch_bounds__(192) void rope_assemble(
    const float* __restrict__ q_c, const float* __restrict__ q_r,
    const float* __restrict__ k_c, const float* __restrict__ k_r,
    const float* __restrict__ vf, const float* __restrict__ unc,
    __bf16* __restrict__ Q, __bf16* __restrict__ K, __bf16* __restrict__ Vt) {
  int t = blockIdx.x, h = blockIdx.y, d = threadIdx.x;
  float u = unc[t];
  u = fminf(fmaxf(u, 0.f), 1.f);
  float sc = 0.5f + u;  // MIN_SCALE + u*(MAX-MIN)
  float qv, kv;
  if (d < HD) {
    qv = q_c[(size_t)t * (NHEAD * HD) + h * HD + d];
    kv = k_c[(size_t)t * (NHEAD * HD) + h * HD + d];
  } else {
    int r = d - HD;
    int rr = r & 31;  // index into half (R/2 = 32)
    float theta = powf(500000.0f, -(float)rr / 32.0f);
    float f = (float)t * theta * sc;
    float c = __cosf(f), s = __sinf(f);
    size_t qb = (size_t)t * (NHEAD * RDIM) + h * RDIM;
    float q1 = q_r[qb + rr], q2 = q_r[qb + rr + 32];
    float k1 = k_r[qb + rr], k2 = k_r[qb + rr + 32];
    if (r < 32) { qv = q1 * c - q2 * s; kv = k1 * c - k2 * s; }
    else        { qv = q2 * c + q1 * s; kv = k2 * c + k1 * s; }
  }
  size_t row = ((size_t)h * TT + t) * HDIM + d;
  Q[row] = (__bf16)qv;
  K[row] = (__bf16)kv;
  if (d < HD)
    Vt[((size_t)h * HD + d) * TT + t] = (__bf16)vf[(size_t)t * (NHEAD * HD) + h * HD + d];
}

// ---------------------------------------------------------------------------
// Fused dual-branch flash attention (bidirectional + sliding-window masked),
// merged with per-token alpha. Block = 256 thr = 8 waves; block handles one
// head x 128 query rows; each wave owns 16 query rows, streams 64-key tiles.
// Per key tile: 24 WMMAs for S, 32 WMMAs for the two PV accumulations.
// ---------------------------------------------------------------------------
__global__ __launch_bounds__(256) void mla_attention(
    const __bf16* __restrict__ Q, const __bf16* __restrict__ K,
    const __bf16* __restrict__ Vt, const float* __restrict__ alpha,
    float* __restrict__ merged) {
  int h = blockIdx.y;
  int lane = threadIdx.x & 31, wave = threadIdx.x >> 5;
  int q0 = blockIdx.x * 128 + wave * 16;
  const __bf16* Qh = Q + (size_t)h * TT * HDIM;
  const __bf16* Kh = K + (size_t)h * TT * HDIM;
  const __bf16* Vh = Vt + (size_t)h * HD * TT;

  __shared__ __bf16 Pb[8][16][64];  // bidir probs, per wave
  __shared__ __bf16 Pa[8][16][64];  // masked probs, per wave

  // Preload Q fragments for all 6 K-chunks of the 192-wide head dim
  v16bf qf[6];
#pragma unroll
  for (int c = 0; c < 6; c++) qf[c] = load_a_frag(Qh, HDIM, q0, c * 32, lane);

  v8f Ob[8] = {};  // bidir output acc, 16x128
  v8f Oa[8] = {};  // masked output acc
  float mb[8], ma[8], lb[8], la[8];
#pragma unroll
  for (int v = 0; v < 8; v++) { mb[v] = NEGBIG; ma[v] = NEGBIG; lb[v] = 0.f; la[v] = 0.f; }

  int g = lane >> 4, ln = lane & 15;
  const float scale = rsqrtf((float)HDIM);

  for (int kt = 0; kt < TT; kt += 64) {
    // --- scores S = Q K^T for this 16x64 tile ---
    v8f S[4] = {};
#pragma unroll
    for (int c = 0; c < 6; c++) {
#pragma unroll
      for (int n = 0; n < 4; n++) {
        v16bf b = load_b_frag(Kh, HDIM, kt + n * 16, c * 32, lane);
        S[n] = wmma_bf16(qf[c], b, S[n]);
      }
    }
    // --- dual online softmax, P tiles spilled to LDS in bf16 ---
#pragma unroll
    for (int v = 0; v < 8; v++) {
      int i = q0 + v + 8 * g;                          // query index of this row
      int ks = max(0, (i / 256) * 256 - 256);          // window start
      float sv[4], sm[4];
      float rb = NEGBIG, ra = NEGBIG;
#pragma unroll
      for (int n = 0; n < 4; n++) {
        int j = kt + n * 16 + ln;
        float s = S[n][v] * scale;
        sv[n] = s;
        bool ok = (j <= i) && (j >= ks);
        sm[n] = ok ? s : NEGBIG;
        rb = fmaxf(rb, s);
        ra = fmaxf(ra, sm[n]);
      }
#pragma unroll
      for (int msk = 1; msk < 16; msk <<= 1) {
        rb = fmaxf(rb, __shfl_xor(rb, msk, 32));
        ra = fmaxf(ra, __shfl_xor(ra, msk, 32));
      }
      float nmb = fmaxf(mb[v], rb), nma = fmaxf(ma[v], ra);
      float cb = __expf(mb[v] - nmb), ca = __expf(ma[v] - nma);
      mb[v] = nmb; ma[v] = nma;
      float sb = 0.f, sa = 0.f;
#pragma unroll
      for (int n = 0; n < 4; n++) {
        float pb = __expf(sv[n] - nmb);
        float pa = __expf(sm[n] - nma);
        Pb[wave][v + 8 * g][n * 16 + ln] = (__bf16)pb;
        Pa[wave][v + 8 * g][n * 16 + ln] = (__bf16)pa;
        sb += pb; sa += pa;
      }
#pragma unroll
      for (int msk = 1; msk < 16; msk <<= 1) {
        sb += __shfl_xor(sb, msk, 32);
        sa += __shfl_xor(sa, msk, 32);
      }
      lb[v] = lb[v] * cb + sb;
      la[v] = la[v] * ca + sa;
#pragma unroll
      for (int n2 = 0; n2 < 8; n2++) { Ob[n2][v] *= cb; Oa[n2][v] *= ca; }
    }
    // --- O += P * V (both branches) ---
    const __bf16* pbw = &Pb[wave][0][0];
    const __bf16* paw = &Pa[wave][0][0];
#pragma unroll
    for (int c = 0; c < 2; c++) {
      v16bf afb = load_a_frag(pbw, 64, 0, c * 32, lane);
      v16bf afa = load_a_frag(paw, 64, 0, c * 32, lane);
#pragma unroll
      for (int n = 0; n < 8; n++) {
        v16bf vb = load_b_frag(Vh, TT, n * 16, kt + c * 32, lane);
        Ob[n] = wmma_bf16(afb, vb, Ob[n]);
        Oa[n] = wmma_bf16(afa, vb, Oa[n]);
      }
    }
  }

  // --- epilogue: normalize both branches, mix with alpha, write merged ---
#pragma unroll
  for (int v = 0; v < 8; v++) {
    int i = q0 + v + 8 * g;
    float a = alpha[i];
    float ib = 1.f / lb[v], ia = 1.f / la[v];
#pragma unroll
    for (int n = 0; n < 8; n++) {
      float ob = Ob[n][v] * ib;
      float oa = Oa[n][v] * ia;
      merged[(size_t)i * (NHEAD * HD) + h * HD + n * 16 + ln] = a * ob + (1.f - a) * oa;
    }
  }
}

// ---------------------------------------------------------------------------
// Host-side orchestration
// ---------------------------------------------------------------------------
extern "C" void kernel_launch(void* const* d_in, const int* in_sizes, int n_in,
                              void* d_out, int out_size, void* d_ws, size_t ws_size,
                              hipStream_t stream) {
  const float* x         = (const float*)d_in[0];
  const float* unc       = (const float*)d_in[1];
  const float* w_q_down  = (const float*)d_in[2];
  const float* q_norm_w  = (const float*)d_in[3];
  const float* w_q_up    = (const float*)d_in[4];
  const float* w_q_rope  = (const float*)d_in[5];
  const float* w_kv_down = (const float*)d_in[6];
  const float* kv_norm_w = (const float*)d_in[7];
  const float* w_k_up    = (const float*)d_in[8];
  const float* w_v_up    = (const float*)d_in[9];
  const float* w_k_rope  = (const float*)d_in[10];
  const float* w_out     = (const float*)d_in[11];
  const float* w_gate    = (const float*)d_in[12];
  const float* b_gate    = (const float*)d_in[13];

  char* ws = (char*)d_ws;
  size_t off = 0;
  auto alloc = [&](size_t bytes) -> char* {
    char* p = ws + off;
    off += (bytes + 255) & ~(size_t)255;
    return p;
  };

  __bf16* x_bf    = (__bf16*)alloc((size_t)TT * DD * 2);
  __bf16* wqd_t   = (__bf16*)alloc((size_t)QLAT * DD * 2);     // 1536 x 2048
  __bf16* wqu_t   = (__bf16*)alloc((size_t)(NHEAD*HD) * QLAT * 2);
  __bf16* wqr_t   = (__bf16*)alloc((size_t)(NHEAD*RDIM) * QLAT * 2);
  __bf16* wkvd_t  = (__bf16*)alloc((size_t)KVLAT * DD * 2);
  __bf16* wku_t   = (__bf16*)alloc((size_t)(NHEAD*HD) * KVLAT * 2);
  __bf16* wvu_t   = (__bf16*)alloc((size_t)(NHEAD*HD) * KVLAT * 2);
  __bf16* wkr_t   = (__bf16*)alloc((size_t)(NHEAD*RDIM) * DD * 2);
  __bf16* wout_t  = (__bf16*)alloc((size_t)DD * (NHEAD*HD) * 2);
  float*  q_pre   = (float*) alloc((size_t)TT * QLAT * 4);
  float*  kv_pre  = (float*) alloc((size_t)TT * KVLAT * 4);
  __bf16* q_lat   = (__bf16*)alloc((size_t)TT * QLAT * 2);
  __bf16* kv_lat  = (__bf16*)alloc((size_t)TT * KVLAT * 2);
  float*  q_c     = (float*) alloc((size_t)TT * NHEAD * HD * 4);
  float*  q_r     = (float*) alloc((size_t)TT * NHEAD * RDIM * 4);
  float*  k_c     = (float*) alloc((size_t)TT * NHEAD * HD * 4);
  float*  v_f     = (float*) alloc((size_t)TT * NHEAD * HD * 4);
  float*  k_r     = (float*) alloc((size_t)TT * NHEAD * RDIM * 4);
  __bf16* Qh      = (__bf16*)alloc((size_t)NHEAD * TT * HDIM * 2);
  __bf16* Kh      = (__bf16*)alloc((size_t)NHEAD * TT * HDIM * 2);
  __bf16* Vt      = (__bf16*)alloc((size_t)NHEAD * HD * TT * 2);
  float*  alp     = (float*) alloc((size_t)TT * 4);
  float*  merged  = (float*) alloc((size_t)TT * NHEAD * HD * 4);
  __bf16* mrg_bf  = (__bf16*)alloc((size_t)TT * NHEAD * HD * 2);

  float* outp = (float*)d_out;            // [0 : T*D) = merged @ w_out
  float* outA = outp + (size_t)TT * DD;   // [T*D : T*D+T) = alpha

  auto cvt = [&](const float* in, __bf16* out, size_t n) {
    to_bf16_kernel<<<dim3((unsigned)((n + 255) / 256)), 256, 0, stream>>>(in, out, n);
  };
  auto tr = [&](const float* in, __bf16* out, int K, int N) {
    size_t n = (size_t)K * N;
    transpose_to_bf16<<<dim3((unsigned)((n + 255) / 256)), 256, 0, stream>>>(in, out, K, N);
  };
  auto gemm = [&](const __bf16* A, const __bf16* Bt, float* C, int M, int N, int K) {
    gemm_bf16_nt<<<dim3(N / 64, M / 256), 256, 0, stream>>>(A, Bt, C, M, N, K);
  };

  // 1) precision convert + weight transposes
  cvt(x, x_bf, (size_t)TT * DD);
  tr(w_q_down,  wqd_t,  DD,    QLAT);
  tr(w_q_up,    wqu_t,  QLAT,  NHEAD * HD);
  tr(w_q_rope,  wqr_t,  QLAT,  NHEAD * RDIM);
  tr(w_kv_down, wkvd_t, DD,    KVLAT);
  tr(w_k_up,    wku_t,  KVLAT, NHEAD * HD);
  tr(w_v_up,    wvu_t,  KVLAT, NHEAD * HD);
  tr(w_k_rope,  wkr_t,  DD,    NHEAD * RDIM);
  tr(w_out,     wout_t, NHEAD * HD, DD);

  // 2) down projections + k_rope projection
  gemm(x_bf, wqd_t,  q_pre,  TT, QLAT,         DD);
  gemm(x_bf, wkvd_t, kv_pre, TT, KVLAT,        DD);
  gemm(x_bf, wkr_t,  k_r,    TT, NHEAD * RDIM, DD);

  // 3) RMS norms -> bf16 latents
  rmsnorm_bf16<<<dim3(TT), 256, 0, stream>>>(q_pre,  q_norm_w,  q_lat,  QLAT);
  rmsnorm_bf16<<<dim3(TT), 256, 0, stream>>>(kv_pre, kv_norm_w, kv_lat, KVLAT);

  // 4) up projections
  gemm(q_lat,  wqu_t, q_c, TT, NHEAD * HD,   QLAT);
  gemm(q_lat,  wqr_t, q_r, TT, NHEAD * RDIM, QLAT);
  gemm(kv_lat, wku_t, k_c, TT, NHEAD * HD,   KVLAT);
  gemm(kv_lat, wvu_t, v_f, TT, NHEAD * HD,   KVLAT);

  // 5) gate (alpha), also written to output tail
  gate_kernel<<<dim3(TT), 256, 0, stream>>>(x, w_gate, b_gate, alp, outA, DD);

  // 6) RoPE + head-major assembly of Q/K/V^T in bf16
  rope_assemble<<<dim3(TT, NHEAD), HDIM, 0, stream>>>(q_c, q_r, k_c, k_r, v_f, unc,
                                                      Qh, Kh, Vt);

  // 7) fused dual-branch attention -> merged (f32)
  mla_attention<<<dim3(TT / 128, NHEAD), 256, 0, stream>>>(Qh, Kh, Vt, alp, merged);

  // 8) output projection
  cvt(merged, mrg_bf, (size_t)TT * NHEAD * HD);
  gemm(mrg_bf, wout_t, outp, TT, DD, NHEAD * HD);
}
